// single_scale_rpn_outputs_63307817943122
// MI455X (gfx1250) — matrix operand order repaired and checked by reading, not compiled, not run on hardware
//
#include <hip/hip_runtime.h>

// ---------------------------------------------------------------------------
// RPN head for MI455X (gfx1250, wave32, WMMA, TDM).
//  GEMM1: implicit im2col 3x3 conv, M=8400 N=1024 K=9216, bf16 WMMA 16x16x32.
//         x repacked NHWC so K-order is (r,s,c); double-buffered LDS with a
//         software-pipelined A prefetch (one barrier per K-tile), B tile moved
//         global->LDS by the Tensor Data Mover (wave 0) when available.
//  GEMM2: fused 1x1 cls+bbox conv, M=8400 N=96(pad of 90) K=1024.
//  Then: sigmoid scores + box decode/clip, 2-level radix threshold for top-2000,
//  stable compaction + bitonic sort (4096 u64 keys in LDS), greedy NMS, top-300.
// ---------------------------------------------------------------------------

typedef __attribute__((ext_vector_type(16))) __bf16 v16bf;
typedef __attribute__((ext_vector_type(8)))  __bf16 bf16x8;
typedef __attribute__((ext_vector_type(8)))  float  v8f;
typedef __attribute__((ext_vector_type(4)))  unsigned uint32x4;
typedef __attribute__((ext_vector_type(8)))  int      int32x8;
typedef __attribute__((ext_vector_type(4)))  int      int32x4;

#if defined(__has_builtin)
# if __has_builtin(__builtin_amdgcn_tensor_load_to_lds) && __has_builtin(__builtin_amdgcn_s_wait_tensorcnt)
#  define USE_TDM 1
# endif
#endif
#ifndef USE_TDM
# define USE_TDM 0
#endif

#define LDA 40            // LDS row stride (bf16 elems): 80B rows -> 16B aligned frags
#define M_TOT 8400        // 2*50*84
#define K1 9216           // 1024*9
#define KT1 288           // K1/32
#define K2 1024
#define KT2 32
#define NPROP 63000       // 50*84*15
#define CAND 4096
#define CLS_OFF 0
#define BBOX_OFF 252000
#define ROIS_OFF 756000
#define PROBS_OFF 759000
#define BBOX_CLIP 4.135166556742356f

__constant__ float ANC[15][4] = {
  {-15.f,-4.f,30.f,19.f},{-38.f,-16.f,53.f,31.f},{-84.f,-40.f,99.f,55.f},
  {-176.f,-88.f,191.f,103.f},{-360.f,-184.f,375.f,199.f},
  {-8.f,-8.f,23.f,23.f},{-24.f,-24.f,39.f,39.f},{-56.f,-56.f,71.f,71.f},
  {-120.f,-120.f,135.f,135.f},{-248.f,-248.f,263.f,263.f},
  {-3.f,-14.f,18.f,29.f},{-14.f,-36.f,29.f,51.f},{-36.f,-80.f,51.f,95.f},
  {-80.f,-168.f,95.f,183.f},{-168.f,-344.f,183.f,359.f}};

// ---------------- pack / convert kernels ----------------
// x: NCHW fp32 -> NHWC bf16  (out index = p*1024 + c, p = b*4200 + y*84 + x)
__global__ void rpn_pack_x(const float* __restrict__ x, __bf16* __restrict__ xb) {
  int i = blockIdx.x * 256 + threadIdx.x;
  if (i >= 8601600) return;
  int c = i & 1023, p = i >> 10;
  int b = p / 4200, sp = p - b * 4200;
  xb[i] = (__bf16)x[(size_t)b * 4300800 + (size_t)c * 4200 + sp];
}

// conv_w: OIHW fp32 -> [n][(r*3+s)*1024 + c] bf16
__global__ void rpn_pack_w1(const float* __restrict__ w, __bf16* __restrict__ w1b) {
  int i = blockIdx.x * 256 + threadIdx.x;
  if (i >= 9437184) return;
  int n = i / K1, t = i - n * K1;
  int rs = t >> 10, c = t & 1023;
  w1b[i] = (__bf16)w[(size_t)n * K1 + c * 9 + rs];
}

__global__ void rpn_pack_w2(const float* __restrict__ cls_w, const float* __restrict__ bbox_w,
                            __bf16* __restrict__ w2b) {
  int i = blockIdx.x * 256 + threadIdx.x;
  if (i >= 96 * 1024) return;
  int n = i >> 10, k = i & 1023;
  float v = 0.f;
  if (n < 30) v = cls_w[n * 1024 + k];
  else if (n < 90) v = bbox_w[(n - 30) * 1024 + k];
  w2b[i] = (__bf16)v;
}

__global__ void rpn_zero_u32(unsigned* __restrict__ p, int n) {
  int i = blockIdx.x * 256 + threadIdx.x;
  if (i < n) p[i] = 0u;
}

// im2col A fetch: one pixel, 16 contiguous NHWC channels, branch-free padding
static __device__ __forceinline__ void g1_loadA(const __bf16* __restrict__ xb, int kt,
                                                bool mvalid, int bimg, int yy, int xx,
                                                int fa_kh, bf16x8& v0, bf16x8& v1) {
  const int rs = kt >> 5;                 // tap index r*3+s (uniform)
  const int c0 = (kt & 31) * 32;          // channel offset (uniform)
  const int r = rs / 3, s = rs - r * 3;
  int iy = yy + r - 1, ix = xx + s - 1;
  bool ok = mvalid && ((unsigned)iy < 50u) && ((unsigned)ix < 84u);
  int iyc = ok ? iy : 0, ixc = ok ? ix : 0;
  const bf16x8* p =
      (const bf16x8*)&xb[(((size_t)bimg * 50 + iyc) * 84 + ixc) * 1024 + c0 + fa_kh];
  v0 = p[0]; v1 = p[1];
  if (!ok) { v0 = (bf16x8){}; v1 = (bf16x8){}; }
}

// ---------------- GEMM1: 3x3 conv (implicit im2col NHWC), bf16 WMMA ----------------
__global__ __launch_bounds__(256) void rpn_gemm1(const __bf16* __restrict__ xb,
                                                 const __bf16* __restrict__ w1b,
                                                 const float* __restrict__ conv_b,
                                                 __bf16* __restrict__ hb) {
  __shared__ __bf16 As[2][128 * LDA];
  __shared__ __bf16 Bs[2][64 * LDA];
  const int tid  = threadIdx.x;
  const int lane = tid & 31;
  const int wv   = tid >> 5;
  const int mtile = blockIdx.x * 128;
  const int ntile = blockIdx.y * 64;
  const int m_off = (wv >> 1) * 32;
  const int n_off = (wv & 1) * 32;

  v8f acc[2][2] = {};

  // A-tile fill mapping: thread -> (row, 16-elem k half); pixel decoded once
  const int fa_m  = tid >> 1;
  const int fa_kh = (tid & 1) * 16;
  const int gm    = mtile + fa_m;
  const bool mvalid = gm < M_TOT;
  int bimg = 0, yy = 0, xx = 0;
  if (mvalid) { bimg = gm / 4200; int sp = gm - bimg * 4200; yy = sp / 84; xx = sp - yy * 84; }
#if !USE_TDM
  const int fb_n = tid >> 2;
  const int fb_q = (tid & 3) * 8;
#endif
  const int hh  = lane >> 4;
  const int l15 = lane & 15;

#if USE_TDM
  // D# group1 is loop-invariant: data_size=2B, pad 4 dwords after every 16 dwords
  // (LDS row stride 80B = LDA), tensor 32 x 64 (dim0 stride 9216), tile 32 x 64.
  const int32x8 g1 = { (int)((1u << 16) | (1u << 20) | (3u << 22) | (3u << 25)),
                       (int)(32u << 16),                 // tensor_dim0 = 32
                       (int)(64u << 16),                 // tensor_dim1 = 64
                       (int)(32u << 16),                 // tile_dim0 = 32
                       (int)64,                          // tile_dim1 = 64
                       (int)9216,                        // tensor_dim0_stride
                       0, 0 };
  const int32x4 gz = { 0, 0, 0, 0 };
  const unsigned ldsB0 = (unsigned)(unsigned long long)(void*)&Bs[0][0];
  const unsigned ldsB1 = (unsigned)(unsigned long long)(void*)&Bs[1][0];
  const unsigned long long wbase =
      (unsigned long long)(const void*)&w1b[(size_t)ntile * K1];
#endif

  // pipeline prologue: stage tile 0 in registers
  bf16x8 a0, a1;
  g1_loadA(xb, 0, mvalid, bimg, yy, xx, fa_kh, a0, a1);
#if !USE_TDM
  bf16x8 breg = *(const bf16x8*)&w1b[(size_t)(ntile + fb_n) * K1 + fb_q];
#endif

  for (int kt = 0; kt < KT1; ++kt) {
    const int pb = kt & 1;
    // --- commit staged tile kt into LDS buffer pb ---
    *(bf16x8*)&As[pb][fa_m * LDA + fa_kh]     = a0;
    *(bf16x8*)&As[pb][fa_m * LDA + fa_kh + 8] = a1;
#if USE_TDM
    if (wv == 0) {
      unsigned long long ga = wbase + (unsigned long long)(kt * 32) * 2ull;
      uint32x4 g0 = { 1u,                                  // count=1, user mode
                      pb ? ldsB1 : ldsB0,
                      (unsigned)(ga & 0xFFFFFFFFull),
                      (unsigned)((ga >> 32) & 0x01FFFFFFull) | 0x80000000u }; // type=2
#  if __clang_major__ >= 23
      __builtin_amdgcn_tensor_load_to_lds(g0, g1, gz, gz, (int32x8){}, 0);
#  else
      __builtin_amdgcn_tensor_load_to_lds(g0, g1, gz, gz, 0);
#  endif
      __builtin_amdgcn_s_wait_tensorcnt(0);
    }
#else
    *(bf16x8*)&Bs[pb][fb_n * LDA + fb_q] = breg;
#endif
    // --- issue global loads for tile kt+1 (consumed next iteration) ---
    if (kt + 1 < KT1) {
      g1_loadA(xb, kt + 1, mvalid, bimg, yy, xx, fa_kh, a0, a1);
#if !USE_TDM
      breg = *(const bf16x8*)&w1b[(size_t)(ntile + fb_n) * K1 + (kt + 1) * 32 + fb_q];
#endif
    }
    __syncthreads();   // buffer pb now valid for all waves

    // --- fragment loads, contiguous k order: lane half hh holds k = 16*hh..16*hh+15 of A and B ---
    v16bf afr[2], bfr[2];
    #pragma unroll
    for (int i = 0; i < 2; ++i) {
      const __bf16* ap = &As[pb][(m_off + i * 16 + l15) * LDA + hh * 16];
      bf16x8 alo = *(const bf16x8*)ap;
      bf16x8 ahi = *(const bf16x8*)(ap + 8);
      afr[i] = __builtin_shufflevector(alo, ahi, 0,1,2,3,4,5,6,7,8,9,10,11,12,13,14,15);
      const __bf16* bp = &Bs[pb][(n_off + i * 16 + l15) * LDA + hh * 16];
      bf16x8 blo = *(const bf16x8*)bp;
      bf16x8 bhi = *(const bf16x8*)(bp + 8);
      bfr[i] = __builtin_shufflevector(blo, bhi, 0,1,2,3,4,5,6,7,8,9,10,11,12,13,14,15);
    }
    #pragma unroll
    for (int i = 0; i < 2; ++i)
      #pragma unroll
      for (int jn = 0; jn < 2; ++jn)
        acc[i][jn] = __builtin_amdgcn_wmma_f32_16x16x32_bf16(
            false, afr[i], false, bfr[jn], (short)0, acc[i][jn], false, false);
    // no trailing barrier: next iteration writes the other buffer, and this
    // wave's s_wait_dscnt before the WMMAs fences its reads of buffer pb.
  }

  // --- epilogue: bias + relu -> h as bf16 in [m][c] ---
  #pragma unroll
  for (int jn = 0; jn < 2; ++jn) {
    const int n = ntile + n_off + jn * 16 + l15;
    const float bias = conv_b[n];
    #pragma unroll
    for (int i = 0; i < 2; ++i) {
      #pragma unroll
      for (int r = 0; r < 8; ++r) {
        int m = mtile + m_off + i * 16 + hh * 8 + r;
        if (m < M_TOT) {
          float v = acc[i][jn][r] + bias;
          v = v > 0.f ? v : 0.f;
          hb[(size_t)m * 1024 + n] = (__bf16)v;
        }
      }
    }
  }
}

// ---------------- GEMM2: fused 1x1 cls(30) + bbox(60) conv ----------------
__global__ __launch_bounds__(256) void rpn_gemm2(const __bf16* __restrict__ hb,
                                                 const __bf16* __restrict__ w2b,
                                                 const float* __restrict__ cls_b,
                                                 const float* __restrict__ bbox_b,
                                                 float* __restrict__ out) {
  __shared__ __bf16 As[128 * LDA];
  __shared__ __bf16 Bs[96 * LDA];
  const int tid  = threadIdx.x;
  const int lane = tid & 31;
  const int wv   = tid >> 5;
  const int mtile = blockIdx.x * 128;
  const int m_off = (wv >> 1) * 32;
  const int n_off = (wv & 1) * 48;

  v8f acc[2][3] = {};

  const int fa_m  = tid >> 1;
  const int fa_kh = (tid & 1) * 16;
  const int gm    = mtile + fa_m;
  const bool mvalid = gm < M_TOT;
  const int hh  = lane >> 4;
  const int l15 = lane & 15;

  for (int kt = 0; kt < KT2; ++kt) {
    const int k0 = kt * 32;
    bf16x8 v0 = {}, v1 = {};
    if (mvalid) {
      v0 = *(const bf16x8*)&hb[(size_t)gm * 1024 + k0 + fa_kh];
      v1 = *(const bf16x8*)&hb[(size_t)gm * 1024 + k0 + fa_kh + 8];
    }
    *(bf16x8*)&As[fa_m * LDA + fa_kh]     = v0;
    *(bf16x8*)&As[fa_m * LDA + fa_kh + 8] = v1;
    for (int u = tid; u < 96 * 4; u += 256) {
      int n = u >> 2, q = (u & 3) * 8;
      *(bf16x8*)&Bs[n * LDA + q] = *(const bf16x8*)&w2b[n * K2 + k0 + q];
    }
    __syncthreads();

    v16bf afr[2], bfr[3];
    #pragma unroll
    for (int i = 0; i < 2; ++i) {
      const __bf16* ap = &As[(m_off + i * 16 + l15) * LDA + hh * 16];
      bf16x8 alo = *(const bf16x8*)ap;
      bf16x8 ahi = *(const bf16x8*)(ap + 8);
      afr[i] = __builtin_shufflevector(alo, ahi, 0,1,2,3,4,5,6,7,8,9,10,11,12,13,14,15);
    }
    #pragma unroll
    for (int jn = 0; jn < 3; ++jn) {
      const __bf16* bp = &Bs[(n_off + jn * 16 + l15) * LDA + hh * 16];
      bf16x8 blo = *(const bf16x8*)bp;
      bf16x8 bhi = *(const bf16x8*)(bp + 8);
      bfr[jn] = __builtin_shufflevector(blo, bhi, 0,1,2,3,4,5,6,7,8,9,10,11,12,13,14,15);
    }
    #pragma unroll
    for (int i = 0; i < 2; ++i)
      #pragma unroll
      for (int jn = 0; jn < 3; ++jn)
        acc[i][jn] = __builtin_amdgcn_wmma_f32_16x16x32_bf16(
            false, afr[i], false, bfr[jn], (short)0, acc[i][jn], false, false);
    __syncthreads();
  }

  #pragma unroll
  for (int jn = 0; jn < 3; ++jn) {
    const int n = n_off + jn * 16 + l15;      // 0..95
    if (n < 90) {
      const float bias = (n < 30) ? cls_b[n] : bbox_b[n - 30];
      #pragma unroll
      for (int i = 0; i < 2; ++i) {
        #pragma unroll
        for (int r = 0; r < 8; ++r) {
          int m = mtile + m_off + i * 16 + hh * 8 + r;
          if (m < M_TOT) {
            int b = m / 4200; int sp = m - b * 4200;
            float v = acc[i][jn][r] + bias;
            if (n < 30) out[CLS_OFF + b * 126000 + n * 4200 + sp] = v;
            else        out[BBOX_OFF + b * 252000 + (n - 30) * 4200 + sp] = v;
          }
        }
      }
    }
  }
}

// ---------------- scores + box decode/clip + level-1 histogram ----------------
__global__ void rpn_score_box(const float* __restrict__ out, const float* __restrict__ im_info,
                              float* __restrict__ scores, float* __restrict__ boxes,
                              unsigned* __restrict__ hist1) {
  int t = blockIdx.x * 256 + threadIdx.x;
  if (t >= 2 * NPROP) return;
  int b = t / NPROP; int i = t - b * NPROP;
  int sp = i / 15; int a = i - sp * 15; int y = sp / 84; int x = sp - y * 84;
  const float* cls = out + CLS_OFF;
  const float* bbp = out + BBOX_OFF;
  float l0 = cls[b * 126000 + a * 4200 + sp];
  float l1 = cls[b * 126000 + (15 + a) * 4200 + sp];
  float sc = 1.f / (1.f + __expf(l0 - l1));   // softmax over 2 classes == sigmoid(l1-l0)
  float d0 = bbp[b * 252000 + (a * 4 + 0) * 4200 + sp];
  float d1 = bbp[b * 252000 + (a * 4 + 1) * 4200 + sp];
  float d2 = bbp[b * 252000 + (a * 4 + 2) * 4200 + sp];
  float d3 = bbp[b * 252000 + (a * 4 + 3) * 4200 + sp];
  float ax1 = ANC[a][0] + x * 16.f, ay1 = ANC[a][1] + y * 16.f;
  float ax2 = ANC[a][2] + x * 16.f, ay2 = ANC[a][3] + y * 16.f;
  float w = ax2 - ax1 + 1.f, h = ay2 - ay1 + 1.f;
  float cx = ax1 + 0.5f * w, cy = ay1 + 0.5f * h;
  float dw = fminf(d2, BBOX_CLIP), dh = fminf(d3, BBOX_CLIP);
  float pcx = d0 * w + cx, pcy = d1 * h + cy;
  float pw = __expf(dw) * w, ph = __expf(dh) * h;
  float x1 = pcx - 0.5f * pw, y1 = pcy - 0.5f * ph;
  float x2 = pcx + 0.5f * pw - 1.f, y2 = pcy + 0.5f * ph - 1.f;
  float hmax = im_info[b * 3 + 0] - 1.f, wmax = im_info[b * 3 + 1] - 1.f;
  x1 = fminf(fmaxf(x1, 0.f), wmax); x2 = fminf(fmaxf(x2, 0.f), wmax);
  y1 = fminf(fmaxf(y1, 0.f), hmax); y2 = fminf(fmaxf(y2, 0.f), hmax);
  scores[t] = sc;
  boxes[(size_t)t * 4 + 0] = x1; boxes[(size_t)t * 4 + 1] = y1;
  boxes[(size_t)t * 4 + 2] = x2; boxes[(size_t)t * 4 + 3] = y2;
  unsigned key = __float_as_uint(sc);           // positive floats order as uints
  atomicAdd(&hist1[b * 1024 + (key >> 22)], 1u);
}

__global__ void rpn_find_t1(const unsigned* __restrict__ hist1, unsigned* __restrict__ t1c1) {
  if (threadIdx.x != 0) return;
  int b = blockIdx.x;
  unsigned acc = 0; int bin = 1023;
  for (; bin >= 0; --bin) {
    unsigned c = hist1[b * 1024 + bin];
    if (acc + c >= 2000u) break;
    acc += c;
  }
  if (bin < 0) bin = 0;
  t1c1[b * 2 + 0] = (unsigned)bin;
  t1c1[b * 2 + 1] = acc;                 // count strictly above bin
}

__global__ void rpn_hist2(const float* __restrict__ scores, const unsigned* __restrict__ t1c1,
                          unsigned* __restrict__ hist2) {
  int t = blockIdx.x * 256 + threadIdx.x;
  if (t >= 2 * NPROP) return;
  int b = t / NPROP;
  unsigned key = __float_as_uint(scores[t]);
  if ((key >> 22) == t1c1[b * 2])
    atomicAdd(&hist2[b * 4096 + ((key >> 10) & 4095u)], 1u);
}

__global__ void rpn_find_t2(const unsigned* __restrict__ hist2, const unsigned* __restrict__ t1c1,
                            unsigned* __restrict__ th22) {
  if (threadIdx.x != 0) return;
  int b = blockIdx.x;
  unsigned acc = t1c1[b * 2 + 1]; int bin = 4095;
  for (; bin >= 0; --bin) {
    unsigned c = hist2[b * 4096 + bin];
    if (acc + c >= 2000u) break;
    acc += c;
  }
  if (bin < 0) bin = 0;
  th22[b] = (t1c1[b * 2] << 12) | (unsigned)bin;   // 22-bit key threshold
}

// ---------------- top-2000 compact + sort + NMS + top-300 emit ----------------
__global__ __launch_bounds__(1024) void rpn_topk_nms(const float* __restrict__ scores,
                                                     const float* __restrict__ boxes,
                                                     const unsigned* __restrict__ th22,
                                                     float* __restrict__ out) {
  __shared__ unsigned long long skey[CAND];  // 32 KB; later reused for box coords
  __shared__ unsigned wsum[32];
  __shared__ unsigned candCount;
  __shared__ unsigned keepw[64];
  __shared__ int supFlag;
  __shared__ unsigned topIdx[2000];
  __shared__ float topSc[2000];

  const int b = blockIdx.x;
  const int tid = threadIdx.x;
  const int lane = tid & 31;
  const int wvi = tid >> 5;
  const unsigned TH = th22[b];

  if (tid == 0) candCount = 0;
  for (int i = tid; i < CAND; i += 1024) skey[i] = ~0ull;
  __syncthreads();

  // ---- stable (index-ordered) compaction of candidates >= threshold ----
  for (int base = 0; base < NPROP; base += 1024) {
    int i = base + tid;
    bool pred = false; unsigned key = 0;
    if (i < NPROP) {
      key = __float_as_uint(scores[b * NPROP + i]);
      pred = ((key >> 10) >= TH);
    }
    unsigned long long mask = __ballot(pred);
    if (lane == 0) wsum[wvi] = (unsigned)__popcll(mask);
    __syncthreads();
    unsigned wbase = 0, total = 0;
    for (int wq = 0; wq < 32; ++wq) { if (wq == wvi) wbase = total; total += wsum[wq]; }
    unsigned cc = candCount;
    unsigned rank = (unsigned)__popcll(mask & ((1ull << lane) - 1ull));
    if (pred) {
      unsigned pos = cc + wbase + rank;
      if (pos < CAND)
        skey[pos] = (((unsigned long long)(~key)) << 32) | (unsigned long long)(unsigned)i;
    }
    __syncthreads();
    if (tid == 0) candCount = cc + total;
    __syncthreads();
  }

  // ---- bitonic sort ascending (~key desc-score, idx asc tiebreak) ----
  for (int k2 = 2; k2 <= CAND; k2 <<= 1) {
    for (int j = k2 >> 1; j > 0; j >>= 1) {
      __syncthreads();
      for (int i = tid; i < CAND; i += 1024) {
        int l = i ^ j;
        if (l > i) {
          unsigned long long A = skey[i], B = skey[l];
          bool up = ((i & k2) == 0);
          if ((A > B) == up) { skey[i] = B; skey[l] = A; }
        }
      }
    }
  }
  __syncthreads();

  // ---- extract top 2000 ----
  for (int i = tid; i < 2000; i += 1024) {
    unsigned long long s = skey[i];
    topIdx[i] = (unsigned)(s & 0xFFFFFFFFull);
    topSc[i]  = __uint_as_float(~(unsigned)(s >> 32));
  }
  __syncthreads();

  // ---- overlay boxes onto skey storage (SoA: x1,y1,x2,y2 at stride 2048) ----
  float* bxs = (float*)skey;
  for (int i = tid; i < 2000; i += 1024) {
    unsigned gi = topIdx[i];
    if (gi >= (unsigned)NPROP) gi = 0;
    const float* bp = &boxes[((size_t)b * NPROP + gi) * 4];
    bxs[i] = bp[0]; bxs[2048 + i] = bp[1]; bxs[4096 + i] = bp[2]; bxs[6144 + i] = bp[3];
  }
  if (tid < 64) keepw[tid] = 0u;
  __syncthreads();

  // ---- greedy NMS (score-sorted order), iou > 0.7 suppresses ----
  for (int i = 0; i < 2000; ++i) {
    if (tid == 0) supFlag = 0;
    __syncthreads();
    float xi1 = bxs[i], yi1 = bxs[2048 + i], xi2 = bxs[4096 + i], yi2 = bxs[6144 + i];
    float areai = (xi2 - xi1 + 1.f) * (yi2 - yi1 + 1.f);
    for (int jj = tid; jj < i; jj += 1024) {
      if (keepw[jj >> 5] & (1u << (jj & 31))) {
        float xx1 = fmaxf(xi1, bxs[jj]);
        float yy1 = fmaxf(yi1, bxs[2048 + jj]);
        float xx2 = fminf(xi2, bxs[4096 + jj]);
        float yy2 = fminf(yi2, bxs[6144 + jj]);
        float inter = fmaxf(xx2 - xx1 + 1.f, 0.f) * fmaxf(yy2 - yy1 + 1.f, 0.f);
        float areaj = (bxs[4096 + jj] - bxs[jj] + 1.f) * (bxs[6144 + jj] - bxs[2048 + jj] + 1.f);
        float iou = inter / (areai + areaj - inter);
        if (iou > 0.7f) supFlag = 1;   // benign same-value race
      }
    }
    __syncthreads();
    if (tid == 0 && !supFlag) keepw[i >> 5] |= (1u << (i & 31));
    __syncthreads();
  }

  // ---- emit rois (600x5) and probs (600x1); invalid slots zeroed ----
  float* rois  = out + ROIS_OFF;
  float* probs = out + PROBS_OFF;
  for (int p = tid; p < 300; p += 1024) {
    float* rr = &rois[(size_t)(b * 300 + p) * 5];
    rr[0] = (float)b; rr[1] = 0.f; rr[2] = 0.f; rr[3] = 0.f; rr[4] = 0.f;
    probs[b * 300 + p] = 0.f;
  }
  __syncthreads();
  for (int i = tid; i < 2000; i += 1024) {
    if (keepw[i >> 5] & (1u << (i & 31))) {
      int r = 0;
      for (int wq = 0; wq < (i >> 5); ++wq) r += __popc(keepw[wq]);
      r += __popc(keepw[i >> 5] & ((1u << (i & 31)) - 1u));
      if (r < 300) {
        float* rr = &rois[(size_t)(b * 300 + r) * 5];
        rr[1] = bxs[i]; rr[2] = bxs[2048 + i]; rr[3] = bxs[4096 + i]; rr[4] = bxs[6144 + i];
        probs[b * 300 + r] = topSc[i];
      }
    }
  }
}

// ---------------------------------------------------------------------------
extern "C" void kernel_launch(void* const* d_in, const int* in_sizes, int n_in,
                              void* d_out, int out_size, void* d_ws, size_t ws_size,
                              hipStream_t stream) {
  const float* x      = (const float*)d_in[0];
  const float* conv_w = (const float*)d_in[1];
  const float* conv_b = (const float*)d_in[2];
  const float* cls_w  = (const float*)d_in[3];
  const float* cls_b  = (const float*)d_in[4];
  const float* bbox_w = (const float*)d_in[5];
  const float* bbox_b = (const float*)d_in[6];
  const float* iminfo = (const float*)d_in[7];
  float* out = (float*)d_out;

  char* ws = (char*)d_ws;
  size_t off = 0;
  auto alloc = [&](size_t bytes) -> void* {
    void* p = ws + off;
    off = (off + bytes + 255) & ~size_t(255);
    return p;
  };
  __bf16* xb   = (__bf16*)alloc((size_t)8601600 * 2);
  __bf16* w1b  = (__bf16*)alloc((size_t)9437184 * 2);
  __bf16* hb   = (__bf16*)alloc((size_t)M_TOT * 1024 * 2);
  __bf16* w2b  = (__bf16*)alloc((size_t)96 * 1024 * 2);
  float*  scrs = (float*)alloc((size_t)2 * NPROP * 4);
  float*  bxs  = (float*)alloc((size_t)2 * NPROP * 16);
  unsigned* hist = (unsigned*)alloc((size_t)(2048 + 8192) * 4); // hist1 | hist2
  unsigned* t1c1 = (unsigned*)alloc(4 * 4);
  unsigned* th22 = (unsigned*)alloc(2 * 4);

  rpn_pack_x<<<(8601600 + 255) / 256, 256, 0, stream>>>(x, xb);
  rpn_pack_w1<<<(9437184 + 255) / 256, 256, 0, stream>>>(conv_w, w1b);
  rpn_pack_w2<<<(98304 + 255) / 256, 256, 0, stream>>>(cls_w, bbox_w, w2b);
  rpn_zero_u32<<<(10240 + 255) / 256, 256, 0, stream>>>(hist, 10240);

  rpn_gemm1<<<dim3(66, 16), 256, 0, stream>>>(xb, w1b, conv_b, hb);
  rpn_gemm2<<<66, 256, 0, stream>>>(hb, w2b, cls_b, bbox_b, out);

  rpn_score_box<<<(2 * NPROP + 255) / 256, 256, 0, stream>>>(out, iminfo, scrs, bxs, hist);
  rpn_find_t1<<<2, 32, 0, stream>>>(hist, t1c1);
  rpn_hist2<<<(2 * NPROP + 255) / 256, 256, 0, stream>>>(scrs, t1c1, hist + 2048);
  rpn_find_t2<<<2, 32, 0, stream>>>(hist + 2048, t1c1, th22);
  rpn_topk_nms<<<2, 1024, 0, stream>>>(scrs, bxs, th22, out);
}
